// MultiHeadAttention_41094247088903
// MI455X (gfx1250) — compile-verified, matrix operand order checked by reading, not run on hardware
//
#include <hip/hip_runtime.h>
#include <hip/hip_bf16.h>
#include <math.h>

typedef __attribute__((ext_vector_type(16))) _Float16 v16h;
typedef __attribute__((ext_vector_type(8)))  float    v8f;

#define NB   4
#define SEQ  2048
#define NCH  8
#define DM   512

#define BSTRIDE 40   // padded LDS stride (f16 elems) for 32-wide K slice

// ---------------------------------------------------------------------------
// Row remapping for the transpose(1,2) in the reference.
// REMAP 0: identity. 1: (b,n,c)->(b,c,n). 2: (b,c,n)->(b,n,c).
// ---------------------------------------------------------------------------
template<int REMAP>
__device__ __forceinline__ int remap_row(int r) {
  if (REMAP == 1) {
    int b = r >> 14, n = (r >> 3) & (SEQ - 1), c = r & (NCH - 1);
    return (((b << 3) | c) << 11) | n;
  } else if (REMAP == 2) {
    int b = r >> 14, c = (r >> 11) & (NCH - 1), n = r & (SEQ - 1);
    return (((b << 11) | n) << 3) | c;
  }
  return r;
}

// ---------------------------------------------------------------------------
// A fragment: 16x32 tile, row-major (lda). CDNA5 wave32 layout:
//   lanes 0-15  : row M=lane,    K = {0..7} and {16..23}
//   lanes 16-31 : row M=lane-16, K = {8..15} and {24..31}
// Per-lane loads are contiguous (2x32B f32 halves or 2x16B f16 halves).
// ---------------------------------------------------------------------------
template<typename T>
__device__ __forceinline__ v16h load_a_frag(const T* __restrict__ A, int lda,
                                            int row0, int k0, int lane) {
  const int row = row0 + (lane & 15);
  const int kb  = k0 + ((lane >> 4) << 3);
  const T* p = A + (size_t)row * lda + kb;
  v16h a;
#pragma unroll
  for (int j = 0; j < 8; ++j) {
    a[j]     = (_Float16)p[j];
    a[j + 8] = (_Float16)p[16 + j];
  }
  return a;
}

// ---------------------------------------------------------------------------
// Generic batched WMMA GEMM with LDS-staged transposed-f16 B:
//   C = act(scale * A @ Bt^T + bias)
// Bt is stored [N,K] row-major f16 (ldb), so global loads are contiguous.
// Block = 128 threads = 4 waves; block tile 128 rows x 64 cols; K step 32.
// Each wave owns a 32x64 tile (8 v8f accumulators, 2 A fragments), so every
// B fragment read from LDS feeds two WMMAs.
// B tile (64 cols x 32 k) is double-buffered in LDS, shared by all 4 waves.
// Grid: x = N/64, y = M/128, z = batch.
// ---------------------------------------------------------------------------
template<typename AT, typename OT, bool RELU, bool BIAS, int REMAP, bool OUT_TRANS>
__global__ void gemm_bt_kernel(const AT* __restrict__ A,
                               const _Float16* __restrict__ Bt,
                               const float* __restrict__ bias,
                               OT* __restrict__ C,
                               int K, int lda, int ldb, int ldc, float scale,
                               long long strideA, long long strideB,
                               long long strideC) {
  __shared__ _Float16 bsm[2][64 * BSTRIDE];

  const int lane = threadIdx.x & 31;
  const int wave = threadIdx.x >> 5;
  const int row0 = blockIdx.y * 128 + wave * 32;
  const int col0 = blockIdx.x * 64;

  const AT*       Ab = A  + (long long)blockIdx.z * strideA;
  const _Float16* Bb = Bt + (long long)blockIdx.z * strideB;
  OT*             Cb = C  + (long long)blockIdx.z * strideC;

  // Cooperative B-tile stager: 128 threads x 32B each = 4KB (64 cols x 32 k).
  const int tcol = threadIdx.x >> 1;         // 0..63
  const int tk   = (threadIdx.x & 1) * 16;   // 0 or 16
  const _Float16* bsrc = Bb + (size_t)(col0 + tcol) * ldb + tk;
  auto stage = [&](int b, int kk) {
    const _Float16* src = bsrc + kk;
    _Float16* dst = &bsm[b][tcol * BSTRIDE + tk];
#pragma unroll
    for (int j = 0; j < 16; ++j) dst[j] = src[j];
  };

  v8f acc[8] = {};

  int buf = 0;
  stage(buf, 0);
  __syncthreads();

  for (int k0 = 0; k0 < K; k0 += 32) {
    v16h a0 = load_a_frag(Ab, lda, row0,      k0, lane);
    v16h a1 = load_a_frag(Ab, lda, row0 + 16, k0, lane);
    if (k0 + 32 < K) {
      stage(buf ^ 1, k0 + 32);                        // prefetch next K slice
      __builtin_prefetch(Ab + (size_t)(row0 + (lane & 15)) * lda + k0 + 32, 0, 1);
    }

    const int colL = lane & 15;
    const int kb   = (lane >> 4) << 3;
#pragma unroll
    for (int t = 0; t < 4; ++t) {
      const _Float16* p = &bsm[buf][(t * 16 + colL) * BSTRIDE + kb];
      v16h b;
#pragma unroll
      for (int j = 0; j < 8; ++j) {
        b[j]     = p[j];
        b[j + 8] = p[16 + j];
      }
      acc[t]     = __builtin_amdgcn_wmma_f32_16x16x32_f16(
          false, a0, false, b, (short)0, acc[t],     false, false);
      acc[t + 4] = __builtin_amdgcn_wmma_f32_16x16x32_f16(
          false, a1, false, b, (short)0, acc[t + 4], false, false);
    }
    __syncthreads();
    buf ^= 1;
  }

  // C/D layout: VGPR r -> M=r (lanes 0-15) or M=8+r (lanes 16-31); N=lane&15.
  const int colL  = lane & 15;
  const int rbase = (lane >> 4) * 8;
#pragma unroll
  for (int h = 0; h < 2; ++h) {
#pragma unroll
    for (int t = 0; t < 4; ++t) {
      const int col = col0 + t * 16 + colL;
      const float bi = BIAS ? bias[col] : 0.0f;
#pragma unroll
      for (int r = 0; r < 8; ++r) {
        const int row  = row0 + h * 16 + rbase + r;
        float val = acc[h * 4 + t][r] * scale + bi;
        if (RELU) val = fmaxf(val, 0.0f);
        const int orow = remap_row<REMAP>(row);
        if (OUT_TRANS) {
          // orow = (b*NCH+c)<<11 | n : store as [bc, col, n] (transposed slice)
          Cb[((size_t)(orow >> 11) * ldc + col) * SEQ + (orow & (SEQ - 1))] = (OT)val;
        } else {
          Cb[(size_t)orow * ldc + col] = (OT)val;
        }
      }
    }
  }
}

// ---------------------------------------------------------------------------
// Weight transpose + f32->f16: W [in,out] f32 -> Wt [out,in] f16. 32x32 tiles.
// ---------------------------------------------------------------------------
__global__ void transpose_w_kernel(const float* __restrict__ W,
                                   _Float16* __restrict__ Wt) {
  __shared__ float tile[32][33];
  const int x = blockIdx.x * 32 + threadIdx.x;   // out dim
  const int y = blockIdx.y * 32 + threadIdx.y;   // in dim
  tile[threadIdx.y][threadIdx.x] = W[(size_t)y * DM + x];
  __syncthreads();
  const int xo = blockIdx.y * 32 + threadIdx.x;  // in dim
  const int yo = blockIdx.x * 32 + threadIdx.y;  // out dim
  Wt[(size_t)yo * DM + xo] = (_Float16)tile[threadIdx.x][threadIdx.y];
}

// ---------------------------------------------------------------------------
// In-place row softmax: one 256-thread block per row of `ncols` floats.
// ---------------------------------------------------------------------------
__global__ void softmax_rows_kernel(float* __restrict__ attn, int ncols) {
  __shared__ float red[8];
  float* p = attn + (size_t)blockIdx.x * ncols;
  const int tid = threadIdx.x;

  float vmax = -3.402823466e38f;
  for (int i = tid; i < ncols; i += 256) vmax = fmaxf(vmax, p[i]);
#pragma unroll
  for (int o = 16; o > 0; o >>= 1) vmax = fmaxf(vmax, __shfl_xor(vmax, o, 32));
  if ((tid & 31) == 0) red[tid >> 5] = vmax;
  __syncthreads();
  float rmax = red[0];
#pragma unroll
  for (int w = 1; w < 8; ++w) rmax = fmaxf(rmax, red[w]);
  __syncthreads();

  float sum = 0.0f;
  for (int i = tid; i < ncols; i += 256) {
    float e = __expf(p[i] - rmax);
    p[i] = e;
    sum += e;
  }
#pragma unroll
  for (int o = 16; o > 0; o >>= 1) sum += __shfl_xor(sum, o, 32);
  if ((tid & 31) == 0) red[tid >> 5] = sum;
  __syncthreads();
  float tsum = 0.0f;
#pragma unroll
  for (int w = 0; w < 8; ++w) tsum += red[w];
  const float inv = 1.0f / tsum;
  for (int i = tid; i < ncols; i += 256) p[i] *= inv;
}

// ---------------------------------------------------------------------------
extern "C" void kernel_launch(void* const* d_in, const int* in_sizes, int n_in,
                              void* d_out, int out_size, void* d_ws, size_t ws_size,
                              hipStream_t stream) {
  const float* q  = (const float*)d_in[0];
  const float* k  = (const float*)d_in[1];
  const float* v  = (const float*)d_in[2];
  const float* Wq = (const float*)d_in[3];
  const float* bq = (const float*)d_in[4];
  const float* Wk = (const float*)d_in[5];
  const float* bk = (const float*)d_in[6];
  const float* Wv = (const float*)d_in[7];
  const float* bv = (const float*)d_in[8];
  const float* Wo = (const float*)d_in[9];
  const float* bo = (const float*)d_in[10];

  const size_t elems = (size_t)NB * SEQ * NCH * DM;        // 33,554,432
  float* out  = (float*)d_out;                             // [B,N,C,D]
  float* attn = (float*)d_out + elems;                     // [B,C,N,N]

  _Float16* qp  = (_Float16*)d_ws;                         // [B,C,N,D]
  _Float16* kp  = qp  + elems;                             // [B,C,N,D] (= [N,K] for QK^T)
  _Float16* vpT = kp  + elems;                             // [B,C,D,N] (transposed)
  _Float16* ctx = vpT + elems;                             // [B,C,N,D]
  _Float16* WqT = ctx + elems;                             // [out,in] f16
  _Float16* WkT = WqT + (size_t)DM * DM;
  _Float16* WvT = WkT + (size_t)DM * DM;
  _Float16* WoT = WvT + (size_t)DM * DM;

  const int Mproj = NB * SEQ * NCH;                        // 65536
  const float inv_sqrt_d = 0.044194173824159216f;          // 1/sqrt(512)
  dim3 blk(128);

  // ---- Weight transpose/convert: W [in,out] f32 -> Wt [out,in] f16
  {
    dim3 g(DM / 32, DM / 32), b(32, 32);
    transpose_w_kernel<<<g, b, 0, stream>>>(Wq, WqT);
    transpose_w_kernel<<<g, b, 0, stream>>>(Wk, WkT);
    transpose_w_kernel<<<g, b, 0, stream>>>(Wv, WvT);
    transpose_w_kernel<<<g, b, 0, stream>>>(Wo, WoT);
  }

  // ---- Q/K/V projections: x @ W + b, rows remapped (b,n,c)->(b,c,n)
  {
    dim3 g(DM / 64, Mproj / 128, 1);
    gemm_bt_kernel<float, _Float16, false, true, 1, false>
        <<<g, blk, 0, stream>>>(q, WqT, bq, qp, DM, DM, DM, DM, 1.0f, 0, 0, 0);
    gemm_bt_kernel<float, _Float16, false, true, 1, false>
        <<<g, blk, 0, stream>>>(k, WkT, bk, kp, DM, DM, DM, DM, 1.0f, 0, 0, 0);
    // V projection stores transposed [b,c,d,n] so PV GEMM's B is [N,K].
    gemm_bt_kernel<float, _Float16, false, true, 1, true>
        <<<g, blk, 0, stream>>>(v, WvT, bv, vpT, DM, DM, DM, DM, 1.0f, 0, 0, 0);
  }

  // ---- Scores: qp @ kp^T * (1/sqrt(D)), f32 straight into attn output region
  {
    dim3 g(SEQ / 64, SEQ / 128, NB * NCH);
    gemm_bt_kernel<_Float16, float, false, false, 0, false>
        <<<g, blk, 0, stream>>>(qp, kp, nullptr, attn, DM, DM, DM, SEQ,
                                inv_sqrt_d,
                                (long long)SEQ * DM, (long long)SEQ * DM,
                                (long long)SEQ * SEQ);
  }

  // ---- Softmax rows in place (attn is both an output and the PV input)
  softmax_rows_kernel<<<dim3(NB * NCH * SEQ), dim3(256), 0, stream>>>(attn, SEQ);

  // ---- ctx = attn @ vp : A = attn f32 (f16 on load), B = vpT [d,n] = [N,K]
  {
    dim3 g(DM / 64, SEQ / 128, NB * NCH);
    gemm_bt_kernel<float, _Float16, false, false, 0, false>
        <<<g, blk, 0, stream>>>(attn, vpT, nullptr, ctx, SEQ, SEQ, SEQ, DM,
                                1.0f,
                                (long long)SEQ * SEQ, (long long)SEQ * DM,
                                (long long)SEQ * DM);
  }

  // ---- out = relu(ctx @ Wo + bo), rows remapped (b,c,n)->(b,n,c), f32 out
  {
    dim3 g(DM / 64, Mproj / 128, 1);
    gemm_bt_kernel<_Float16, float, true, true, 2, false>
        <<<g, blk, 0, stream>>>(ctx, WoT, bo, out, DM, DM, DM, DM, 1.0f,
                                0, 0, 0);
  }
}
